// GNet_32152125178026
// MI455X (gfx1250) — matrix operand order repaired and checked by reading, not compiled, not run on hardware
//
#include <hip/hip_runtime.h>
#include <hip/hip_bf16.h>
#include <math.h>

// ---------------------------------------------------------------------------
// Problem constants (match reference)
// ---------------------------------------------------------------------------
#define N_DETS   8192
#define DEG      64
#define E_EDGES  (N_DETS * DEG)     // 524288
#define NUM_BLOCKS 8
#define WAVES    8                  // 256-thread blocks = 8 wave32

typedef __attribute__((ext_vector_type(16))) __bf16          v16bf;
typedef __attribute__((ext_vector_type(2)))  __bf16          v2bf;
typedef __attribute__((ext_vector_type(8)))  float           v8f;
typedef __attribute__((ext_vector_type(8)))  unsigned short  us8;
typedef __attribute__((ext_vector_type(16))) unsigned short  us16;

// round-to-nearest-even fp32 -> bf16 (exact; used in weight prep and raw edge features)
__device__ __forceinline__ unsigned short f2bf(float x) {
  unsigned u = __float_as_uint(x);
  u += 0x7FFFu + ((u >> 16) & 1u);
  return (unsigned short)(u >> 16);
}

// packed fp32x2 -> bf16x2.
// Preferred: hardware v_cvt_pk_bf16_f32. Fallback: round-to-nearest
// (ties away) + v_perm_b32 byte pack = 3 short-latency VALU ops per pair.
__device__ __forceinline__ unsigned cvt_pk_bf16(float a, float b) {
#if __has_builtin(__builtin_amdgcn_cvt_pk_bf16_f32)
  return __builtin_bit_cast(unsigned, __builtin_amdgcn_cvt_pk_bf16_f32(a, b));
#else
  unsigned ua = __float_as_uint(a) + 0x8000u;
  unsigned ub = __float_as_uint(b) + 0x8000u;
  // dst.b0=ua.b2(sel 6) dst.b1=ua.b3(sel 7) dst.b2=ub.b2(sel 2) dst.b3=ub.b3(sel 3)
  return __builtin_amdgcn_perm(ua, ub, 0x03020706u);
#endif
}

__device__ __forceinline__ void store2_bf16(unsigned short* p0, unsigned short* p1,
                                            float a, float b) {
  unsigned d = cvt_pk_bf16(a, b);
  *p0 = (unsigned short)(d & 0xFFFFu);
  *p1 = (unsigned short)(d >> 16);
}

// A-fragment (16x32 bf16): M = lane&15, K runs [8h,8h+8) and [16+8h,24+8h)
__device__ __forceinline__ v16bf loadA_runs(const unsigned short* p) {
  us8 lo = *(const us8*)p;
  us8 hi = *(const us8*)(p + 16);
  us16 t;
#pragma unroll
  for (int i = 0; i < 8; ++i) { t[i] = lo[i]; t[8 + i] = hi[i]; }
  return __builtin_bit_cast(v16bf, t);
}
__device__ __forceinline__ v16bf loadA(const unsigned short* base, int ld, int m, int h) {
  return loadA_runs(base + m * ld + 8 * h);
}
// A fragment from fp32 source (convert on load, packed)
__device__ __forceinline__ v16bf loadA_f32(const float* base, int ld, int m, int h) {
  const float* r = base + m * ld + 8 * h;
  us16 t;
#pragma unroll
  for (int i = 0; i < 8; i += 2) {
    unsigned d0 = cvt_pk_bf16(r[i], r[i + 1]);
    unsigned d1 = cvt_pk_bf16(r[16 + i], r[16 + i + 1]);
    t[i] = (unsigned short)d0; t[i + 1] = (unsigned short)(d0 >> 16);
    t[8 + i] = (unsigned short)d1; t[8 + i + 1] = (unsigned short)(d1 >> 16);
  }
  return __builtin_bit_cast(v16bf, t);
}
// B-fragment from transposed weights Wt[N][Kpad]: lane holds col nt*16+(lane&15),
// K runs [kc*32+8h, +8) and [kc*32+16+8h, +8) -> two 16B reads
__device__ __forceinline__ v16bf loadB(const unsigned short* wt, int kpad,
                                       int nt, int kc, int m, int h) {
  const unsigned short* p = wt + (nt * 16 + m) * kpad + kc * 32 + 8 * h;
  return loadA_runs(p);
}

__device__ __forceinline__ v8f wmma_bf16(v16bf a, v16bf b, v8f c) {
  return __builtin_amdgcn_wmma_f32_16x16x32_bf16(false, a, false, b, (short)0, c,
                                                 false, false);
}

// wave-local LDS ordering (CDNA5 split counters)
__device__ __forceinline__ void wave_lds_fence() {
  __asm__ volatile("s_wait_dscnt 0" ::: "memory");
}

// bias + relu + bf16 store of one 16x16 D tile into row-major staging
__device__ __forceinline__ void store_tile_relu(unsigned short* st, int ldst, int nt,
                                                v8f acc, const float* sbias,
                                                int m, int h) {
  const float b = sbias[nt * 16 + m];
  unsigned short* p = st + 8 * h * ldst + nt * 16 + m;
#pragma unroll
  for (int r = 0; r < 8; r += 2)
    store2_bf16(p + r * ldst, p + (r + 1) * ldst,
                fmaxf(acc[r] + b, 0.f), fmaxf(acc[r + 1] + b, 0.f));
}

// ---------------------------------------------------------------------------
// Weight transpose + bf16 convert: fp32 [K][N] -> bf16 [N][Kpad] (zero pad)
// ---------------------------------------------------------------------------
__global__ __launch_bounds__(256) void wconv_kernel(const float* __restrict__ src,
                                                    unsigned short* __restrict__ dst,
                                                    int K, int N, int Kpad) {
  int i = blockIdx.x * blockDim.x + threadIdx.x;
  if (i >= N * Kpad) return;
  int n = i / Kpad, k = i - n * Kpad;
  float v = (k < K) ? src[(size_t)k * N + n] : 0.f;
  dst[i] = f2bf(v);
}

__global__ __launch_bounds__(256) void copyf_kernel(const float* __restrict__ src,
                                                    float* __restrict__ dst, int n) {
  int i = blockIdx.x * blockDim.x + threadIdx.x;
  if (i < n) dst[i] = src[i];
}

// ---------------------------------------------------------------------------
// Kernel 1: raw pairwise features + MLP 9->256->256->32 (dominant 8e10 FLOP)
// Weights + biases LDS-resident; ping-pong activation staging per wave;
// layer1 runs 4 accumulator tiles per pass to stay within VGPR budget.
// ---------------------------------------------------------------------------
__global__ __launch_bounds__(256) void edge_mlp_kernel(
    const float* __restrict__ detScores, const float* __restrict__ dtBoxes,
    const int* __restrict__ cIdxs, const int* __restrict__ nIdxs,
    const unsigned short* __restrict__ W0t, const float* __restrict__ b0,
    const unsigned short* __restrict__ W1t, const float* __restrict__ b1,
    const unsigned short* __restrict__ W2t, const float* __restrict__ b2,
    unsigned short* __restrict__ pf, int itersPerWave) {
  extern __shared__ __align__(128) unsigned char smem[];
  float* sb0 = (float*)smem;                    // 256
  float* sb1 = sb0 + 256;                       // 256
  float* sb2 = sb1 + 256;                       // 32  (pad to 576 floats)
  unsigned short* sW0    = (unsigned short*)(smem + 576 * 4);  // 256*32
  unsigned short* sW1    = sW0 + 256 * 32;                     // 256*256
  unsigned short* sW2    = sW1 + 256 * 256;                    // 32*256
  unsigned short* rawAll = sW2 + 32 * 256;                     // WAVES*16*32
  unsigned short* stAAll = rawAll + WAVES * 16 * 32;           // WAVES*16*256
  unsigned short* stBAll = stAAll + WAVES * 16 * 256;          // WAVES*16*256

  const int tid = threadIdx.x;
  for (int i = tid; i < 256; i += blockDim.x) { sb0[i] = b0[i]; sb1[i] = b1[i]; }
  for (int i = tid; i < 32; i += blockDim.x) sb2[i] = b2[i];
  for (int i = tid; i < 256 * 32;  i += blockDim.x) sW0[i] = W0t[i];
  for (int i = tid; i < 256 * 256; i += blockDim.x) sW1[i] = W1t[i];
  for (int i = tid; i < 32 * 256;  i += blockDim.x) sW2[i] = W2t[i];
  __syncthreads();

  const int wave = tid >> 5, lane = tid & 31, m = lane & 15, h = lane >> 4;
  unsigned short* raw = rawAll + wave * 16 * 32;
  unsigned short* stA = stAAll + wave * 16 * 256;
  unsigned short* stB = stBAll + wave * 16 * 256;
  if (lane < 16) {
#pragma unroll
    for (int j = 9; j < 32; ++j) raw[m * 32 + j] = 0;   // K padding stays zero
  }

  const int tileBase = blockIdx.x * (itersPerWave * WAVES);
  for (int it = 0; it < itersPerWave; ++it) {
    const int tile = tileBase + it * WAVES + wave;  // grid sized exactly

    // --- raw pairwise features (lanes 0..15, one edge each) ---
    if (lane < 16) {
      const int e = tile * 16 + m;
      const int c = cIdxs[e], n = nIdxs[e];
      const float cx1 = dtBoxes[c * 4 + 0], cy1 = dtBoxes[c * 4 + 1];
      const float cx2 = dtBoxes[c * 4 + 2], cy2 = dtBoxes[c * 4 + 3];
      const float nx1 = dtBoxes[n * 4 + 0], ny1 = dtBoxes[n * 4 + 1];
      const float nx2 = dtBoxes[n * 4 + 2], ny2 = dtBoxes[n * 4 + 3];
      const float cw = cx2 - cx1, ch_ = cy2 - cy1, nw = nx2 - nx1, nh = ny2 - ny1;
      const float areaC = cw * ch_, areaN = nw * nh;
      const float ix = fmaxf(0.f, fminf(cx2, nx2) - fmaxf(cx1, nx1));
      const float iy = fmaxf(0.f, fminf(cy2, ny2) - fmaxf(cy1, ny1));
      const float inter = ix * iy;
      const float iou = inter / (areaC + areaN - inter);
      const float cs = (cw + ch_) * 0.5f;
      const float xd = (nx1 + nw * 0.5f) - (cx1 + cw * 0.5f);
      const float yd = (ny1 + nh * 0.5f) - (cy1 + ch_ * 0.5f);
      const float l2 = sqrtf(xd * xd + yd * yd) / cs;
      raw[m * 32 + 0] = f2bf(detScores[c]);
      raw[m * 32 + 1] = f2bf(detScores[n]);
      raw[m * 32 + 2] = f2bf(iou);
      raw[m * 32 + 3] = f2bf(xd / cs);
      raw[m * 32 + 4] = f2bf(yd / cs);
      raw[m * 32 + 5] = f2bf(l2);
      raw[m * 32 + 6] = f2bf(log2f(nw / cw));
      raw[m * 32 + 7] = f2bf(log2f(nh / ch_));
      raw[m * 32 + 8] = f2bf(log2f(nw / nh) - log2f(cw / ch_));
    }
    wave_lds_fence();

    // --- layer0: 9(pad32) -> 256 : raw -> stA ---
    {
      v16bf a0 = loadA(raw, 32, m, h);
#pragma unroll
      for (int nt = 0; nt < 16; ++nt) {
        v8f acc = {};
        acc = wmma_bf16(a0, loadB(sW0, 32, nt, 0, m, h), acc);
        store_tile_relu(stA, 256, nt, acc, sb0, m, h);
      }
    }
    wave_lds_fence();

    // --- layer1: 256 -> 256 : stA -> stB, 4 N-tiles per pass ---
#pragma unroll
    for (int g = 0; g < 4; ++g) {
      v8f acc0 = {}, acc1 = {}, acc2 = {}, acc3 = {};
#pragma unroll
      for (int k = 0; k < 8; ++k) {
        v16bf a = loadA(stA + k * 32, 256, m, h);
        acc0 = wmma_bf16(a, loadB(sW1, 256, g * 4 + 0, k, m, h), acc0);
        acc1 = wmma_bf16(a, loadB(sW1, 256, g * 4 + 1, k, m, h), acc1);
        acc2 = wmma_bf16(a, loadB(sW1, 256, g * 4 + 2, k, m, h), acc2);
        acc3 = wmma_bf16(a, loadB(sW1, 256, g * 4 + 3, k, m, h), acc3);
      }
      store_tile_relu(stB, 256, g * 4 + 0, acc0, sb1, m, h);
      store_tile_relu(stB, 256, g * 4 + 1, acc1, sb1, m, h);
      store_tile_relu(stB, 256, g * 4 + 2, acc2, sb1, m, h);
      store_tile_relu(stB, 256, g * 4 + 3, acc3, sb1, m, h);
    }
    wave_lds_fence();

    // --- layer2: 256 -> 32 : stB -> pf (global bf16) ---
    {
      v8f acc0 = {}, acc1 = {};
#pragma unroll
      for (int k = 0; k < 8; ++k) {
        v16bf a = loadA(stB + k * 32, 256, m, h);
        acc0 = wmma_bf16(a, loadB(sW2, 256, 0, k, m, h), acc0);
        acc1 = wmma_bf16(a, loadB(sW2, 256, 1, k, m, h), acc1);
      }
      unsigned short* pr = pf + (size_t)(tile * 16 + 8 * h) * 32 + m;
      const float bia0 = sb2[m], bia1 = sb2[16 + m];
#pragma unroll
      for (int r = 0; r < 8; r += 2) {
        store2_bf16(pr + r * 32, pr + (r + 1) * 32,
                    fmaxf(acc0[r] + bia0, 0.f), fmaxf(acc0[r + 1] + bia0, 0.f));
        store2_bf16(pr + r * 32 + 16, pr + (r + 1) * 32 + 16,
                    fmaxf(acc1[r] + bia1, 0.f), fmaxf(acc1[r + 1] + bia1, 0.f));
      }
    }
  }
}

// ---------------------------------------------------------------------------
// Kernel 2: f1 = relu(f @ fc1W + b)  [8192x128]x[128x32] -> bf16 [8192x32]
// ---------------------------------------------------------------------------
__global__ __launch_bounds__(256) void node_fc1_kernel(
    const float* __restrict__ f,
    const unsigned short* __restrict__ fc1t,   // [32][128]
    const float* __restrict__ fc1b,
    unsigned short* __restrict__ f1) {
  extern __shared__ __align__(128) unsigned char smem[];
  float* sb = (float*)smem;                                 // 32 (pad 64)
  unsigned short* sW = (unsigned short*)(smem + 64 * 4);    // 32*128
  const int tid = threadIdx.x;
  for (int i = tid; i < 32; i += blockDim.x) sb[i] = fc1b[i];
  for (int i = tid; i < 32 * 128; i += blockDim.x) sW[i] = fc1t[i];
  __syncthreads();
  const int wave = tid >> 5, lane = tid & 31, m = lane & 15, h = lane >> 4;
  const int tile = blockIdx.x * WAVES + wave;
  v16bf a[4];
#pragma unroll
  for (int k = 0; k < 4; ++k)
    a[k] = loadA_f32(f + (size_t)tile * 16 * 128 + k * 32, 128, m, h);
  v8f acc0 = {}, acc1 = {};
#pragma unroll
  for (int k = 0; k < 4; ++k) {
    acc0 = wmma_bf16(a[k], loadB(sW, 128, 0, k, m, h), acc0);
    acc1 = wmma_bf16(a[k], loadB(sW, 128, 1, k, m, h), acc1);
  }
  unsigned short* pr = f1 + (size_t)(tile * 16 + 8 * h) * 32 + m;
  const float bia0 = sb[m], bia1 = sb[16 + m];
#pragma unroll
  for (int r = 0; r < 8; r += 2) {
    store2_bf16(pr + r * 32, pr + (r + 1) * 32,
                fmaxf(acc0[r] + bia0, 0.f), fmaxf(acc0[r + 1] + bia0, 0.f));
    store2_bf16(pr + r * 32 + 16, pr + (r + 1) * 32 + 16,
                fmaxf(acc1[r] + bia1, 0.f), fmaxf(acc1[r + 1] + bia1, 0.f));
  }
}

// ---------------------------------------------------------------------------
// Kernel 3: edge MLP [pf|f1[c]|f1[n]](96) -> 64 -> 64  + segment_max.
// cIdxs = repeat(arange(N),64) -> one wave per detection, 4 contiguous
// 16-edge tiles, D-fragment row reduce + shfl_xor(16) half-merge.
// ---------------------------------------------------------------------------
__global__ __launch_bounds__(256) void block_edge_kernel(
    const unsigned short* __restrict__ pf, const unsigned short* __restrict__ f1,
    const int* __restrict__ nIdxs,
    const unsigned short* __restrict__ pw1t,  // [64][96]
    const float* __restrict__ pw1b,
    const unsigned short* __restrict__ pw2t,  // [64][64]
    const float* __restrict__ pw2b,
    float* __restrict__ pooled) {
  extern __shared__ __align__(128) unsigned char smem[];
  float* sb1 = (float*)smem;            // 64
  float* sb2 = sb1 + 64;                // 64 -> 128 floats total
  unsigned short* sP1   = (unsigned short*)(smem + 128 * 4);  // 64*96
  unsigned short* sP2   = sP1 + 64 * 96;                      // 64*64
  unsigned short* stAll = sP2 + 64 * 64;                      // WAVES*16*64
  const int tid = threadIdx.x;
  for (int i = tid; i < 64; i += blockDim.x) { sb1[i] = pw1b[i]; sb2[i] = pw2b[i]; }
  for (int i = tid; i < 64 * 96; i += blockDim.x) sP1[i] = pw1t[i];
  for (int i = tid; i < 64 * 64; i += blockDim.x) sP2[i] = pw2t[i];
  __syncthreads();
  const int wave = tid >> 5, lane = tid & 31, m = lane & 15, h = lane >> 4;
  unsigned short* stage = stAll + wave * 16 * 64;
  const int det = blockIdx.x * WAVES + wave;

  float pm[4] = {0.f, 0.f, 0.f, 0.f};  // relu outputs >= 0
#pragma unroll
  for (int t = 0; t < 4; ++t) {
    const int e0 = det * 64 + t * 16;
    v16bf aPF = loadA(pf + (size_t)e0 * 32, 32, m, h);
    v16bf aC  = loadA_runs(f1 + (size_t)det * 32 + 8 * h);   // same row for all M
    const int n = nIdxs[e0 + m];
    v16bf aN  = loadA_runs(f1 + (size_t)n * 32 + 8 * h);
#pragma unroll
    for (int nt = 0; nt < 4; ++nt) {
      v8f acc = {};
      acc = wmma_bf16(aPF, loadB(sP1, 96, nt, 0, m, h), acc);
      acc = wmma_bf16(aC,  loadB(sP1, 96, nt, 1, m, h), acc);
      acc = wmma_bf16(aN,  loadB(sP1, 96, nt, 2, m, h), acc);
      store_tile_relu(stage, 64, nt, acc, sb1, m, h);
    }
    wave_lds_fence();
    v16bf h0 = loadA(stage + 0 * 32, 64, m, h);
    v16bf h1 = loadA(stage + 1 * 32, 64, m, h);
#pragma unroll
    for (int nt = 0; nt < 4; ++nt) {
      v8f acc = {};
      acc = wmma_bf16(h0, loadB(sP2, 64, nt, 0, m, h), acc);
      acc = wmma_bf16(h1, loadB(sP2, 64, nt, 1, m, h), acc);
      const float bias = sb2[nt * 16 + m];
      float mx = 0.f;
#pragma unroll
      for (int r = 0; r < 8; ++r) mx = fmaxf(mx, fmaxf(acc[r] + bias, 0.f));
      pm[nt] = fmaxf(pm[nt], mx);
    }
    wave_lds_fence();
  }
#pragma unroll
  for (int nt = 0; nt < 4; ++nt) {
    float o = __shfl_xor(pm[nt], 16, 32);   // merge lane halves (M rows)
    float v = fmaxf(pm[nt], o);
    if (h == 0) pooled[(size_t)det * 64 + nt * 16 + m] = v;
  }
}

// ---------------------------------------------------------------------------
// Kernel 4: pooled -> post0 -> post1 -> f = relu(f + pooled@outW + b)
// ---------------------------------------------------------------------------
__global__ __launch_bounds__(256) void node_post_kernel(
    const float* __restrict__ pooled,
    const unsigned short* __restrict__ p0t,  // [64][64]
    const float* __restrict__ p0b,
    const unsigned short* __restrict__ p1t,  // [64][64]
    const float* __restrict__ p1b,
    const unsigned short* __restrict__ outt, // [128][64]
    const float* __restrict__ outb,
    float* __restrict__ f) {
  extern __shared__ __align__(128) unsigned char smem[];
  float* sb0 = (float*)smem;            // 64
  float* sb1 = sb0 + 64;                // 64
  float* sbo = sb1 + 64;                // 128 -> 256 floats total
  unsigned short* sP0   = (unsigned short*)(smem + 256 * 4);
  unsigned short* sP1   = sP0 + 64 * 64;
  unsigned short* sO    = sP1 + 64 * 64;       // 128*64
  unsigned short* stAll = sO + 128 * 64;       // WAVES*16*64
  const int tid = threadIdx.x;
  for (int i = tid; i < 64; i += blockDim.x) { sb0[i] = p0b[i]; sb1[i] = p1b[i]; }
  for (int i = tid; i < 128; i += blockDim.x) sbo[i] = outb[i];
  for (int i = tid; i < 64 * 64;  i += blockDim.x) sP0[i] = p0t[i];
  for (int i = tid; i < 64 * 64;  i += blockDim.x) sP1[i] = p1t[i];
  for (int i = tid; i < 128 * 64; i += blockDim.x) sO[i] = outt[i];
  __syncthreads();
  const int wave = tid >> 5, lane = tid & 31, m = lane & 15, h = lane >> 4;
  unsigned short* stage = stAll + wave * 16 * 64;
  const int tile = blockIdx.x * WAVES + wave;

  v16bf a[2];
#pragma unroll
  for (int k = 0; k < 2; ++k)
    a[k] = loadA_f32(pooled + (size_t)tile * 16 * 64 + k * 32, 64, m, h);
#pragma unroll
  for (int nt = 0; nt < 4; ++nt) {           // post0
    v8f acc = {};
    acc = wmma_bf16(a[0], loadB(sP0, 64, nt, 0, m, h), acc);
    acc = wmma_bf16(a[1], loadB(sP0, 64, nt, 1, m, h), acc);
    store_tile_relu(stage, 64, nt, acc, sb0, m, h);
  }
  wave_lds_fence();
#pragma unroll
  for (int k = 0; k < 2; ++k) a[k] = loadA(stage + k * 32, 64, m, h);
#pragma unroll
  for (int nt = 0; nt < 4; ++nt) {           // post1 (overwrites stage)
    v8f acc = {};
    acc = wmma_bf16(a[0], loadB(sP1, 64, nt, 0, m, h), acc);
    acc = wmma_bf16(a[1], loadB(sP1, 64, nt, 1, m, h), acc);
    store_tile_relu(stage, 64, nt, acc, sb1, m, h);
  }
  wave_lds_fence();
#pragma unroll
  for (int k = 0; k < 2; ++k) a[k] = loadA(stage + k * 32, 64, m, h);
#pragma unroll
  for (int nt = 0; nt < 8; ++nt) {           // 64 -> 128 + residual + relu
    v8f acc = {};
    acc = wmma_bf16(a[0], loadB(sO, 64, nt, 0, m, h), acc);
    acc = wmma_bf16(a[1], loadB(sO, 64, nt, 1, m, h), acc);
    const float bias = sbo[nt * 16 + m];
#pragma unroll
    for (int r = 0; r < 8; ++r) {
      const size_t idx = (size_t)(tile * 16 + r + 8 * h) * 128 + nt * 16 + m;
      f[idx] = fmaxf(f[idx] + acc[r] + bias, 0.f);
    }
  }
}

// ---------------------------------------------------------------------------
// Host launcher
// ---------------------------------------------------------------------------
extern "C" void kernel_launch(void* const* d_in, const int* in_sizes, int n_in,
                              void* d_out, int out_size, void* d_ws, size_t ws_size,
                              hipStream_t stream) {
  (void)in_sizes; (void)n_in; (void)out_size; (void)ws_size;

  const float* detScores   = (const float*)d_in[0];
  const float* dtBoxes     = (const float*)d_in[1];
  const float* detFeatures = (const float*)d_in[2];
  const int*   cIdxs       = (const int*)d_in[3];
  const int*   nIdxs       = (const int*)d_in[4];
  const float* pw_W0 = (const float*)d_in[5];
  const float* pw_b0 = (const float*)d_in[6];
  const float* pw_W1 = (const float*)d_in[7];
  const float* pw_b1 = (const float*)d_in[8];
  const float* pw_W2 = (const float*)d_in[9];
  const float* pw_b2 = (const float*)d_in[10];
  const float* blk_fc1_W  = (const float*)d_in[11];
  const float* blk_fc1_b  = (const float*)d_in[12];
  const float* blk_pw1_W  = (const float*)d_in[13];
  const float* blk_pw1_b  = (const float*)d_in[14];
  const float* blk_pw2_W  = (const float*)d_in[15];
  const float* blk_pw2_b  = (const float*)d_in[16];
  const float* blk_post_W = (const float*)d_in[17];
  const float* blk_post_b = (const float*)d_in[18];
  const float* blk_out_W  = (const float*)d_in[19];
  const float* blk_out_b  = (const float*)d_in[20];

  char* ws = (char*)d_ws;
  size_t off = 0;
  auto take = [&](size_t bytes) -> char* {
    char* p = ws + off;
    off += (bytes + 255) & ~(size_t)255;
    return p;
  };
  unsigned short* pf     = (unsigned short*)take((size_t)E_EDGES * 32 * 2);
  float*          f      = (float*)take((size_t)N_DETS * 128 * 4);
  unsigned short* f1     = (unsigned short*)take((size_t)N_DETS * 32 * 2);
  float*          pooled = (float*)take((size_t)N_DETS * 64 * 4);
  unsigned short* W0t = (unsigned short*)take(256 * 32 * 2);
  unsigned short* W1t = (unsigned short*)take(256 * 256 * 2);
  unsigned short* W2t = (unsigned short*)take(32 * 256 * 2);
  unsigned short *fc1t[NUM_BLOCKS], *pw1t[NUM_BLOCKS], *pw2t[NUM_BLOCKS];
  unsigned short *p0t[NUM_BLOCKS], *p1t[NUM_BLOCKS], *outt[NUM_BLOCKS];
  for (int i = 0; i < NUM_BLOCKS; ++i) {
    fc1t[i] = (unsigned short*)take(32 * 128 * 2);
    pw1t[i] = (unsigned short*)take(64 * 96 * 2);
    pw2t[i] = (unsigned short*)take(64 * 64 * 2);
    p0t[i]  = (unsigned short*)take(64 * 64 * 2);
    p1t[i]  = (unsigned short*)take(64 * 64 * 2);
    outt[i] = (unsigned short*)take(128 * 64 * 2);
  }

  auto conv = [&](const float* s, unsigned short* d, int K, int N, int Kpad) {
    int total = N * Kpad;
    wconv_kernel<<<(total + 255) / 256, 256, 0, stream>>>(s, d, K, N, Kpad);
  };
  conv(pw_W0, W0t, 9, 256, 32);
  conv(pw_W1, W1t, 256, 256, 256);
  conv(pw_W2, W2t, 256, 32, 256);
  for (int i = 0; i < NUM_BLOCKS; ++i) {
    conv(blk_fc1_W  + (size_t)i * 128 * 32,          fc1t[i], 128, 32, 128);
    conv(blk_pw1_W  + (size_t)i * 96 * 64,           pw1t[i], 96, 64, 96);
    conv(blk_pw2_W  + (size_t)i * 64 * 64,           pw2t[i], 64, 64, 64);
    conv(blk_post_W + (size_t)(2 * i + 0) * 64 * 64, p0t[i],  64, 64, 64);
    conv(blk_post_W + (size_t)(2 * i + 1) * 64 * 64, p1t[i],  64, 64, 64);
    conv(blk_out_W  + (size_t)i * 64 * 128,          outt[i], 64, 128, 64);
  }
  copyf_kernel<<<(N_DETS * 128 + 255) / 256, 256, 0, stream>>>(detFeatures, f,
                                                               N_DETS * 128);

  // pairwise MLP (dominant compute)
  {
    const int numTiles = E_EDGES / 16;            // 32768
    const int grid = 256;
    const int iters = numTiles / (grid * WAVES);  // 16, exact
    const size_t shb = 576 * 4 +
        (size_t)(256 * 32 + 256 * 256 + 32 * 256 +
                 WAVES * 16 * 32 + WAVES * 16 * 256 + WAVES * 16 * 256) * 2;
    edge_mlp_kernel<<<grid, 256, shb, stream>>>(detScores, dtBoxes, cIdxs, nIdxs,
                                                W0t, pw_b0, W1t, pw_b1, W2t, pw_b2,
                                                pf, iters);
  }

  const size_t sh2 = 64 * 4 + (size_t)(32 * 128) * 2;
  const size_t sh3 = 128 * 4 + (size_t)(64 * 96 + 64 * 64 + WAVES * 16 * 64) * 2;
  const size_t sh4 = 256 * 4 +
      (size_t)(64 * 64 + 64 * 64 + 128 * 64 + WAVES * 16 * 64) * 2;
  for (int i = 0; i < NUM_BLOCKS; ++i) {
    node_fc1_kernel<<<N_DETS / 16 / WAVES, 256, sh2, stream>>>(
        f, fc1t[i], blk_fc1_b + (size_t)i * 32, f1);
    block_edge_kernel<<<N_DETS / WAVES, 256, sh3, stream>>>(
        pf, f1, nIdxs, pw1t[i], blk_pw1_b + (size_t)i * 64, pw2t[i],
        blk_pw2_b + (size_t)i * 64, pooled);
    node_post_kernel<<<N_DETS / 16 / WAVES, 256, sh4, stream>>>(
        pooled, p0t[i], blk_post_b + (size_t)(2 * i + 0) * 64, p1t[i],
        blk_post_b + (size_t)(2 * i + 1) * 64, outt[i], blk_out_b + (size_t)i * 128, f);
  }

  hipMemcpyAsync(d_out, f, (size_t)N_DETS * 128 * sizeof(float),
                 hipMemcpyDeviceToDevice, stream);
}
